// NystromAttention_23373212025390
// MI455X (gfx1250) — compile-verified, matrix operand order checked by reading, not run on hardware
//
#include <hip/hip_runtime.h>
#include <hip/hip_bf16.h>

// ---------------------------------------------------------------------------
// Nystrom attention for MI455X (gfx1250, wave32, WMMA 16x16x32 f16->f32).
// B=4 H=12 S=4096 D=64 L=128 seg=32.
// ---------------------------------------------------------------------------

typedef __attribute__((ext_vector_type(16))) _Float16 v16h;
typedef __attribute__((ext_vector_type(8)))  float    v8f;
typedef _Float16 half_t;

#define NB      4
#define NH      12
#define NHEADS  (NB * NH)      // 48
#define SQ      4096
#define HD      64
#define NL      128
#define SEG     32
#define NSPLIT  8              // key-split factor for the flash stage
#define CHUNK   (SQ / NSPLIT)  // 512 keys per partial
#define NEG_BIG 1000000000.0f
#define SCALE   0.35355339059327373f   // 1/sqrt(sqrt(64))

__device__ __forceinline__ int lane_id() { return (int)(threadIdx.x & 31); }

__device__ __forceinline__ v8f zero8() {
  v8f z;
#pragma unroll
  for (int i = 0; i < 8; ++i) z[i] = 0.0f;
  return z;
}

// K index packing for 16-bit A/B fragments (ISA 7.12.2, 16-bit A 16x32):
// lanes 0-15 : j=0..7 -> K=j,      j=8..15 -> K=16+(j-8)
// lanes 16-31: j=0..7 -> K=8+j,    j=8..15 -> K=24+(j-8)
__device__ __forceinline__ int frag_k(int j, int hi) {
  return (j & 7) + ((j >> 3) << 4) + (hi << 3);
}

// A fragment: 16x32 tile, element (m,k) at p[m*ld + k]
__device__ __forceinline__ v16h load_frag_a(const half_t* p, int ld) {
  int l = lane_id(), m = l & 15, hi = l >> 4;
  v16h a;
#pragma unroll
  for (int j = 0; j < 16; ++j) a[j] = p[m * ld + frag_k(j, hi)];
  return a;
}

// B fragment: 32x16 tile (K x N), element (k,n) at p[k*ld + n]
__device__ __forceinline__ v16h load_frag_b_kxn(const half_t* p, int ld) {
  int l = lane_id(), n = l & 15, hi = l >> 4;
  v16h b;
#pragma unroll
  for (int j = 0; j < 16; ++j) b[j] = p[frag_k(j, hi) * ld + n];
  return b;
}

// B fragment from a row-major N x K source (i.e. computes S = A * src^T):
// element (k,n) at p[n*ld + k]  -- contiguous per lane -> b128 loads
__device__ __forceinline__ v16h load_frag_b_nxk(const half_t* p, int ld) {
  int l = lane_id(), n = l & 15, hi = l >> 4;
  v16h b;
#pragma unroll
  for (int j = 0; j < 16; ++j) b[j] = p[n * ld + frag_k(j, hi)];
  return b;
}

// B fragment of (d*I - M), M tile at p (k x n, ld), global bases kb/nb
__device__ __forceinline__ v16h load_frag_b_diag_minus(const half_t* p, int ld,
                                                       float d, int kb, int nb) {
  int l = lane_id(), n = l & 15, hi = l >> 4;
  v16h b;
#pragma unroll
  for (int j = 0; j < 16; ++j) {
    int k = frag_k(j, hi);
    float v = (((kb + k) == (nb + n)) ? d : 0.0f) - (float)p[k * ld + n];
    b[j] = (half_t)v;
  }
  return b;
}

__device__ __forceinline__ v8f wmma16(v16h a, v16h b, v8f c) {
  return __builtin_amdgcn_wmma_f32_16x16x32_f16(false, a, false, b,
                                                (short)0, c, false, false);
}

// store C fragment (16x16 f32) as f16, row-major ld
__device__ __forceinline__ void store_tile_f16(half_t* p, int ld, int m0, int n0, v8f c) {
  int l = lane_id(), n = l & 15, hi = l >> 4;
#pragma unroll
  for (int i = 0; i < 8; ++i)
    p[(m0 + i + 8 * hi) * ld + n0 + n] = (half_t)c[i];
}

// store (d*I - C) fragment as f16
__device__ __forceinline__ void store_tile_diag_minus(half_t* p, int ld, int m0, int n0,
                                                      v8f c, float d) {
  int l = lane_id(), n = l & 15, hi = l >> 4;
#pragma unroll
  for (int i = 0; i < 8; ++i) {
    int r = m0 + i + 8 * hi, cc = n0 + n;
    p[r * ld + cc] = (half_t)(((r == cc) ? d : 0.0f) - c[i]);
  }
}

__device__ __forceinline__ float red16_max(float v) {
#pragma unroll
  for (int m = 8; m >= 1; m >>= 1) v = fmaxf(v, __shfl_xor(v, m, 32));
  return v;
}
__device__ __forceinline__ float red16_sum(float v) {
#pragma unroll
  for (int m = 8; m >= 1; m >>= 1) v += __shfl_xor(v, m, 32);
  return v;
}

// ---------------------------------------------------------------------------
// K_init: reset the global column-sum max scalar
// ---------------------------------------------------------------------------
__global__ void nys_init(unsigned int* colmax) { *colmax = 0u; }

// ---------------------------------------------------------------------------
// K0: Qh = Q*mask*scale, Kh = K*mask*scale, Vh = V  (f32 -> f16, 4 elem/thread)
// ---------------------------------------------------------------------------
__global__ void nys_prep(const float* __restrict__ Q, const float* __restrict__ K,
                         const float* __restrict__ V, const float* __restrict__ mask,
                         half_t* __restrict__ Qh, half_t* __restrict__ Kh,
                         half_t* __restrict__ Vh) {
  long t   = (long)blockIdx.x * blockDim.x + threadIdx.x;
  long idx = t * 4;                                // 4 consecutive elems, same row
  long bh  = idx / ((long)SQ * HD);
  long rem = idx % ((long)SQ * HD);
  int  s   = (int)(rem / HD);
  int  b   = (int)(bh / NH);
  float m  = mask[(long)b * SQ + s] * SCALE;
  float4 q = *(const float4*)(Q + idx);
  float4 k = *(const float4*)(K + idx);
  float4 v = *(const float4*)(V + idx);
  Qh[idx + 0] = (half_t)(q.x * m); Qh[idx + 1] = (half_t)(q.y * m);
  Qh[idx + 2] = (half_t)(q.z * m); Qh[idx + 3] = (half_t)(q.w * m);
  Kh[idx + 0] = (half_t)(k.x * m); Kh[idx + 1] = (half_t)(k.y * m);
  Kh[idx + 2] = (half_t)(k.z * m); Kh[idx + 3] = (half_t)(k.w * m);
  Vh[idx + 0] = (half_t)v.x;       Vh[idx + 1] = (half_t)v.y;
  Vh[idx + 2] = (half_t)v.z;       Vh[idx + 3] = (half_t)v.w;
}

// ---------------------------------------------------------------------------
// K1: landmark means (mean over 32 consecutive rows of Qs/Ks) -> f16
// grid: NHEADS*NL blocks, HD threads
// ---------------------------------------------------------------------------
__global__ void nys_landmarks(const float* __restrict__ Q, const float* __restrict__ K,
                              const float* __restrict__ mask,
                              half_t* __restrict__ Qlm, half_t* __restrict__ Klm) {
  int bh = blockIdx.x / NL;
  int l  = blockIdx.x % NL;
  int d  = threadIdx.x;
  int b  = bh / NH;
  float accQ = 0.0f, accK = 0.0f;
#pragma unroll 4
  for (int r = 0; r < SEG; ++r) {
    int s   = l * SEG + r;
    float f = mask[(long)b * SQ + s] * SCALE;
    long base = ((long)bh * SQ + s) * HD + d;
    accQ += Q[base] * f;
    accK += K[base] * f;
  }
  long o = ((long)bh * NL + l) * HD + d;
  Qlm[o] = (half_t)(accQ * (1.0f / SEG));
  Klm[o] = (half_t)(accK * (1.0f / SEG));
}

// ---------------------------------------------------------------------------
// K2: kernel_2 = softmax(Qlm @ Klm^T)  (128x128 per head, f16 out)
//     + global max of column sums (for iterative_inv init_scale; row sums == 1)
// grid: NHEADS blocks, 256 threads (8 waves, 16-row bands)
// ---------------------------------------------------------------------------
__global__ void nys_kernel2(const half_t* __restrict__ Qlm, const half_t* __restrict__ Klm,
                            half_t* __restrict__ K2h, unsigned int* __restrict__ colmax) {
  __shared__ float colsum[NL];
  int head = blockIdx.x, tid = threadIdx.x;
  int wave = tid >> 5, lane = tid & 31;
  int n_ = lane & 15, hi = lane >> 4;
  if (tid < NL) colsum[tid] = 0.0f;
  __syncthreads();

  int m0 = wave * 16;
  const half_t* Qb = Qlm + ((long)head * NL + m0) * HD;
  v16h a0 = load_frag_a(Qb, HD);
  v16h a1 = load_frag_a(Qb + 32, HD);

  v8f s[8];
#pragma unroll
  for (int j = 0; j < 8; ++j) {
    const half_t* Kb = Klm + (long)head * NL * HD + (j * 16) * HD;
    v8f c = zero8();
    c = wmma16(a0, load_frag_b_nxk(Kb, HD), c);
    c = wmma16(a1, load_frag_b_nxk(Kb + 32, HD), c);
    s[j] = c;
  }
  float rm[8], rs[8];
#pragma unroll
  for (int i = 0; i < 8; ++i) {
    float mx = -3.0e38f;
#pragma unroll
    for (int j = 0; j < 8; ++j) mx = fmaxf(mx, s[j][i]);
    rm[i] = red16_max(mx);
  }
#pragma unroll
  for (int i = 0; i < 8; ++i) {
    float sm = 0.0f;
#pragma unroll
    for (int j = 0; j < 8; ++j) { s[j][i] = __expf(s[j][i] - rm[i]); sm += s[j][i]; }
    rs[i] = 1.0f / red16_sum(sm);
  }
  half_t* K2b = K2h + (long)head * NL * NL;
#pragma unroll
  for (int j = 0; j < 8; ++j) {
    float cs = 0.0f;
#pragma unroll
    for (int i = 0; i < 8; ++i) {
      float p = s[j][i] * rs[i];
      cs += p;
      K2b[(m0 + i + 8 * hi) * NL + j * 16 + n_] = (half_t)p;
    }
    cs += __shfl_xor(cs, 16, 32);        // add the other half's 8 rows
    if (hi == 0) atomicAdd(&colsum[j * 16 + n_], cs);
  }
  __syncthreads();
  if (tid < NL) atomicMax(colmax, __float_as_uint(colsum[tid]));
}

// ---------------------------------------------------------------------------
// K3: Z = iterative_inv(kernel_2), 6 Newton-Schulz iterations, all in LDS.
//     Vm <- 0.25*Vm @ (13I - KV@(15I - KV@(7I - KV))),  KV = A@Vm
// grid: NHEADS blocks, 256 threads, 128 KB dynamic LDS (4 x 128x128 f16)
// ---------------------------------------------------------------------------
__global__ void nys_invert(const half_t* __restrict__ K2h,
                           const unsigned int* __restrict__ colmax,
                           half_t* __restrict__ Zh) {
  extern __shared__ half_t sm[];
  half_t* Ash  = sm;                 // 128x128
  half_t* Vsh  = sm + NL * NL;       // 128x128
  half_t* KVsh = sm + 2 * NL * NL;   // 128x128
  half_t* Msh  = sm + 3 * NL * NL;   // 128x128

  int head = blockIdx.x, tid = threadIdx.x;
  int wave = tid >> 5;
  int m0 = wave * 16;
  const half_t* A = K2h + (long)head * NL * NL;

  for (int i = tid; i < NL * NL; i += 256) Ash[i] = A[i];
  __syncthreads();
  float init_scale = 1.0f / __uint_as_float(*colmax);   // max row sum == 1
  for (int i = tid; i < NL * NL; i += 256) {
    int r = i >> 7, c = i & 127;
    Vsh[i] = (half_t)(init_scale * (float)Ash[c * NL + r]);   // init = s * A^T
  }
  __syncthreads();

  for (int it = 0; it < 6; ++it) {
    // KV = A @ V
#pragma unroll
    for (int t = 0; t < 8; ++t) {
      v8f acc = zero8();
#pragma unroll
      for (int k0 = 0; k0 < NL; k0 += 32) {
        v16h a = load_frag_a(Ash + m0 * NL + k0, NL);
        v16h b = load_frag_b_kxn(Vsh + k0 * NL + t * 16, NL);
        acc = wmma16(a, b, acc);
      }
      store_tile_f16(KVsh, NL, m0, t * 16, acc);
    }
    __syncthreads();
    // M = 15I - KV @ (7I - KV)
#pragma unroll
    for (int t = 0; t < 8; ++t) {
      v8f acc = zero8();
#pragma unroll
      for (int k0 = 0; k0 < NL; k0 += 32) {
        v16h a = load_frag_a(KVsh + m0 * NL + k0, NL);
        v16h b = load_frag_b_diag_minus(KVsh + k0 * NL + t * 16, NL, 7.0f, k0, t * 16);
        acc = wmma16(a, b, acc);
      }
      store_tile_diag_minus(Msh, NL, m0, t * 16, acc, 15.0f);
    }
    __syncthreads();
    // M <- 13I - KV @ M   (compute into regs, then overwrite Msh)
    v8f r3[8];
#pragma unroll
    for (int t = 0; t < 8; ++t) {
      v8f acc = zero8();
#pragma unroll
      for (int k0 = 0; k0 < NL; k0 += 32) {
        v16h a = load_frag_a(KVsh + m0 * NL + k0, NL);
        v16h b = load_frag_b_kxn(Msh + k0 * NL + t * 16, NL);
        acc = wmma16(a, b, acc);
      }
      r3[t] = acc;
    }
    __syncthreads();
#pragma unroll
    for (int t = 0; t < 8; ++t) store_tile_diag_minus(Msh, NL, m0, t * 16, r3[t], 13.0f);
    __syncthreads();
    // V <- 0.25 * V @ M   (regs, then overwrite Vsh)
    v8f rv[8];
#pragma unroll
    for (int t = 0; t < 8; ++t) {
      v8f acc = zero8();
#pragma unroll
      for (int k0 = 0; k0 < NL; k0 += 32) {
        v16h a = load_frag_a(Vsh + m0 * NL + k0, NL);
        v16h b = load_frag_b_kxn(Msh + k0 * NL + t * 16, NL);
        acc = wmma16(a, b, acc);
      }
#pragma unroll
      for (int i = 0; i < 8; ++i) acc[i] *= 0.25f;
      rv[t] = acc;
    }
    __syncthreads();
#pragma unroll
    for (int t = 0; t < 8; ++t) store_tile_f16(Vsh, NL, m0, t * 16, rv[t]);
    __syncthreads();
  }
  half_t* Zb = Zh + (long)head * NL * NL;
  for (int i = tid; i < NL * NL; i += 256) Zb[i] = Vsh[i];
}

// ---------------------------------------------------------------------------
// K4a: partial flash pass of kernel_3 @ V over a 512-key chunk.
//      Stores unnormalized O (128x64), row max, row sum per (head, chunk).
// grid: (NHEADS, NSPLIT), 256 threads (8 waves x 16-row bands).
// No block barriers: the P staging tile is wave-private and same-wave LDS
// ops complete in order (ISA ch.5: DScnt in-order per wave).
// ---------------------------------------------------------------------------
__global__ void nys_fpart(const half_t* __restrict__ Qlm, const half_t* __restrict__ Kh,
                          const half_t* __restrict__ Vh, const float* __restrict__ mask,
                          float* __restrict__ Opart, float* __restrict__ Rm,
                          float* __restrict__ Rs) {
  __shared__ half_t Pl[8][16 * 32];     // wave-private P tiles (16x32)
  int head = blockIdx.x, part = blockIdx.y, b = head / NH;
  int tid = threadIdx.x, wave = tid >> 5, lane = tid & 31;
  int n_ = lane & 15, hi = lane >> 4;
  int m0 = wave * 16;

  const half_t* Qb = Qlm + ((long)head * NL + m0) * HD;
  v16h a0 = load_frag_a(Qb, HD);
  v16h a1 = load_frag_a(Qb + 32, HD);

  float rm[8], rs[8];
  v8f O[4];
#pragma unroll
  for (int i = 0; i < 8; ++i) { rm[i] = -3.0e38f; rs[i] = 0.0f; }
#pragma unroll
  for (int t = 0; t < 4; ++t) O[t] = zero8();

  int nBeg = part * CHUNK, nEnd = nBeg + CHUNK;
  for (int n0 = nBeg; n0 < nEnd; n0 += 32) {
    const half_t* Kb = Kh + ((long)head * SQ + n0) * HD;
    v8f c0 = zero8(), c1 = zero8();
    c0 = wmma16(a0, load_frag_b_nxk(Kb, HD), c0);
    c0 = wmma16(a1, load_frag_b_nxk(Kb + 32, HD), c0);
    const half_t* Kb2 = Kb + 16 * HD;
    c1 = wmma16(a0, load_frag_b_nxk(Kb2, HD), c1);
    c1 = wmma16(a1, load_frag_b_nxk(Kb2 + 32, HD), c1);

    float bias0 = -NEG_BIG * (1.0f - mask[(long)b * SQ + n0 + n_]);
    float bias1 = -NEG_BIG * (1.0f - mask[(long)b * SQ + n0 + 16 + n_]);
#pragma unroll
    for (int i = 0; i < 8; ++i) { c0[i] += bias0; c1[i] += bias1; }

#pragma unroll
    for (int i = 0; i < 8; ++i) {
      float tm = red16_max(fmaxf(c0[i], c1[i]));
      float nm = fmaxf(rm[i], tm);
      float corr = __expf(rm[i] - nm);
      rm[i] = nm;
      rs[i] *= corr;
#pragma unroll
      for (int t = 0; t < 4; ++t) O[t][i] *= corr;
      float p0 = __expf(c0[i] - nm);
      float p1 = __expf(c1[i] - nm);
      rs[i] += red16_sum(p0 + p1);
      int row = i + 8 * hi;
      Pl[wave][row * 32 + n_]      = (half_t)p0;
      Pl[wave][row * 32 + 16 + n_] = (half_t)p1;
    }
    v16h ap = load_frag_a(&Pl[wave][0], 32);
    const half_t* Vb = Vh + ((long)head * SQ + n0) * HD;
#pragma unroll
    for (int t = 0; t < 4; ++t)
      O[t] = wmma16(ap, load_frag_b_kxn(Vb + t * 16, HD), O[t]);
  }
  // write partials
  long base = (long)(head * NSPLIT + part) * NL;
#pragma unroll
  for (int i = 0; i < 8; ++i) {
    int row = m0 + i + 8 * hi;
    if (n_ == 0) { Rm[base + row] = rm[i]; Rs[base + row] = rs[i]; }
#pragma unroll
    for (int t = 0; t < 4; ++t)
      Opart[(base + row) * HD + t * 16 + n_] = O[t][i];
  }
}

// ---------------------------------------------------------------------------
// K4b: combine NSPLIT flash partials -> F (LDS, f16), then W = Z @ F
// grid: NHEADS blocks, 256 threads
// ---------------------------------------------------------------------------
__global__ void nys_fcombine(const float* __restrict__ Opart, const float* __restrict__ Rm,
                             const float* __restrict__ Rs, const half_t* __restrict__ Zh,
                             half_t* __restrict__ Wh) {
  __shared__ float Ssh[NSPLIT][NL];   // per-partial rescale factors
  __shared__ float Dsh[NL];           // 1/denominator per row
  __shared__ half_t Fsh[NL * HD];
  int head = blockIdx.x, tid = threadIdx.x;
  int wave = tid >> 5, lane = tid & 31;
  int n_ = lane & 15, hi = lane >> 4;

  if (tid < NL) {
    float rmv[NSPLIT];
    float M = -3.0e38f;
#pragma unroll
    for (int j = 0; j < NSPLIT; ++j) {
      rmv[j] = Rm[(long)(head * NSPLIT + j) * NL + tid];
      M = fmaxf(M, rmv[j]);
    }
    float d = 0.0f;
#pragma unroll
    for (int j = 0; j < NSPLIT; ++j) {
      float e = __expf(rmv[j] - M);
      Ssh[j][tid] = e;
      d += Rs[(long)(head * NSPLIT + j) * NL + tid] * e;
    }
    Dsh[tid] = 1.0f / d;
  }
  __syncthreads();
  for (int e = tid; e < NL * HD; e += 256) {
    int r = e >> 6, c = e & 63;
    float acc = 0.0f;
#pragma unroll
    for (int j = 0; j < NSPLIT; ++j)
      acc += Opart[((long)(head * NSPLIT + j) * NL + r) * HD + c] * Ssh[j][r];
    Fsh[e] = (half_t)(acc * Dsh[r]);
  }
  __syncthreads();
  // W = Z @ F
  int m0 = wave * 16;
  const half_t* Zb = Zh + (long)head * NL * NL;
  half_t* Wb = Wh + (long)head * NL * HD;
#pragma unroll
  for (int t = 0; t < 4; ++t) {
    v8f acc = zero8();
#pragma unroll
    for (int k0 = 0; k0 < NL; k0 += 32) {
      v16h az = load_frag_a(Zb + m0 * NL + k0, NL);
      v16h bf = load_frag_b_kxn(Fsh + k0 * HD + t * 16, HD);
      acc = wmma16(az, bf, acc);
    }
#pragma unroll
    for (int i = 0; i < 8; ++i)
      Wb[(m0 + i + 8 * hi) * HD + t * 16 + n_] = (half_t)acc[i];
  }
}

// ---------------------------------------------------------------------------
// K5: X = softmax(Qs @ Klm^T) @ W   ->  out f32 (B,H,S,D)
// grid: (SQ/128, NHEADS), 256 threads (8 waves x 16-row bands)
// P staging tile is wave-private -> no block barrier needed.
// ---------------------------------------------------------------------------
__global__ void nys_out(const half_t* __restrict__ Qh, const half_t* __restrict__ Klm,
                        const half_t* __restrict__ Wh, float* __restrict__ out) {
  __shared__ half_t Pl[8][16 * NL];   // wave-private 16x128 P
  int head = blockIdx.y;
  int tid = threadIdx.x, wave = tid >> 5, lane = tid & 31;
  int n_ = lane & 15, hi = lane >> 4;
  int m0 = blockIdx.x * 128 + wave * 16;

  const half_t* Qb = Qh + ((long)head * SQ + m0) * HD;
  v16h a0 = load_frag_a(Qb, HD);
  v16h a1 = load_frag_a(Qb + 32, HD);

  v8f s[8];
#pragma unroll
  for (int j = 0; j < 8; ++j) {
    const half_t* Kb = Klm + (long)head * NL * HD + (j * 16) * HD;
    v8f c = zero8();
    c = wmma16(a0, load_frag_b_nxk(Kb, HD), c);
    c = wmma16(a1, load_frag_b_nxk(Kb + 32, HD), c);
    s[j] = c;
  }
  // full-row softmax over 128 landmarks
#pragma unroll
  for (int i = 0; i < 8; ++i) {
    float mx = -3.0e38f;
#pragma unroll
    for (int j = 0; j < 8; ++j) mx = fmaxf(mx, s[j][i]);
    mx = red16_max(mx);
    float sm = 0.0f;
#pragma unroll
    for (int j = 0; j < 8; ++j) { s[j][i] = __expf(s[j][i] - mx); sm += s[j][i]; }
    float inv = 1.0f / red16_sum(sm);
    int row = i + 8 * hi;
#pragma unroll
    for (int j = 0; j < 8; ++j)
      Pl[wave][row * NL + j * 16 + n_] = (half_t)(s[j][i] * inv);
  }
  // X = P @ W  (same-wave LDS ordering, no barrier)
  const half_t* Wb = Wh + (long)head * NL * HD;
#pragma unroll
  for (int t = 0; t < 4; ++t) {
    v8f acc = zero8();
#pragma unroll
    for (int k0 = 0; k0 < NL; k0 += 32) {
      v16h ap = load_frag_a(&Pl[wave][k0], NL);
      v16h bw = load_frag_b_kxn(Wb + k0 * HD + t * 16, HD);
      acc = wmma16(ap, bw, acc);
    }
#pragma unroll
    for (int i = 0; i < 8; ++i)
      out[((long)head * SQ + m0 + i + 8 * hi) * HD + t * 16 + n_] = acc[i];
  }
}

// ---------------------------------------------------------------------------
extern "C" void kernel_launch(void* const* d_in, const int* in_sizes, int n_in,
                              void* d_out, int out_size, void* d_ws, size_t ws_size,
                              hipStream_t stream) {
  const float* Q    = (const float*)d_in[0];
  const float* K    = (const float*)d_in[1];
  const float* V    = (const float*)d_in[2];
  const float* mask = (const float*)d_in[3];
  float* out        = (float*)d_out;

  const size_t NE = (size_t)NHEADS * SQ * HD;         // 12,582,912
  char* ws = (char*)d_ws;
  size_t off = 0;
  auto alloc = [&](size_t bytes) {
    char* p = ws + off;
    off += (bytes + 255) & ~(size_t)255;
    return p;
  };
  half_t* Qh   = (half_t*)alloc(NE * sizeof(half_t));
  half_t* Kh   = (half_t*)alloc(NE * sizeof(half_t));
  half_t* Vh   = (half_t*)alloc(NE * sizeof(half_t));
  half_t* Qlm  = (half_t*)alloc((size_t)NHEADS * NL * HD * sizeof(half_t));
  half_t* Klm  = (half_t*)alloc((size_t)NHEADS * NL * HD * sizeof(half_t));
  half_t* K2h  = (half_t*)alloc((size_t)NHEADS * NL * NL * sizeof(half_t));
  half_t* Zh   = (half_t*)alloc((size_t)NHEADS * NL * NL * sizeof(half_t));
  half_t* Wh   = (half_t*)alloc((size_t)NHEADS * NL * HD * sizeof(half_t));
  float*  Opart = (float*)alloc((size_t)NHEADS * NSPLIT * NL * HD * sizeof(float));
  float*  Rm    = (float*)alloc((size_t)NHEADS * NSPLIT * NL * sizeof(float));
  float*  Rs    = (float*)alloc((size_t)NHEADS * NSPLIT * NL * sizeof(float));
  unsigned int* colmax = (unsigned int*)alloc(256);

  nys_init<<<1, 1, 0, stream>>>(colmax);
  nys_prep<<<(unsigned)(NE / 4 / 256), 256, 0, stream>>>(Q, K, V, mask, Qh, Kh, Vh);
  nys_landmarks<<<NHEADS * NL, HD, 0, stream>>>(Q, K, mask, Qlm, Klm);
  nys_kernel2<<<NHEADS, 256, 0, stream>>>(Qlm, Klm, K2h, colmax);
  nys_invert<<<NHEADS, 256, 4 * NL * NL * sizeof(half_t), stream>>>(K2h, colmax, Zh);
  nys_fpart<<<dim3(NHEADS, NSPLIT), 256, 0, stream>>>(Qlm, Kh, Vh, mask, Opart, Rm, Rs);
  nys_fcombine<<<NHEADS, 256, 0, stream>>>(Opart, Rm, Rs, Zh, Wh);
  nys_out<<<dim3(SQ / 128, NHEADS), 256, 0, stream>>>(Qh, Klm, Wh, out);
}
